// EncoderGAT_25185688224508
// MI455X (gfx1250) — matrix operand order repaired and checked by reading, not compiled, not run on hardware
//
#include <hip/hip_runtime.h>
#include <hip/hip_bf16.h>

// ---------- types for WMMA ----------
typedef __bf16 bf16_t;
typedef bf16_t v16bf __attribute__((ext_vector_type(16)));
typedef float  v8f   __attribute__((ext_vector_type(8)));
typedef int    v4i_t __attribute__((vector_size(16)));   // 16B vector for async builtin

// ---------- small device helpers ----------
__device__ __forceinline__ unsigned short f2bf(float f) {
    unsigned u = __float_as_uint(f);
    unsigned r = u + 0x7FFFu + ((u >> 16) & 1u);   // round-to-nearest-even
    return (unsigned short)(r >> 16);
}
__device__ __forceinline__ float bf2f(unsigned short h) {
    return __uint_as_float(((unsigned)h) << 16);
}
// monotone order-preserving float<->uint mapping (for atomicMax on floats)
__device__ __forceinline__ unsigned fmap(float f) {
    unsigned u = __float_as_uint(f);
    return (u >> 31) ? ~u : (u | 0x80000000u);
}
__device__ __forceinline__ float funmap(unsigned u) {
    unsigned v = (u >> 31) ? (u & 0x7FFFFFFFu) : ~u;
    return __uint_as_float(v);
}
__device__ __forceinline__ void edge_sd(const int* sA, const int* dA,
                                        long long e, long long E, int& s, int& d) {
    if (e < E) { s = sA[e]; d = dA[e]; }
    else       { int n = (int)(e - E); s = n; d = n; }   // self-loops appended
}

// ---------- CDNA5 async global->LDS copy (ASYNCcnt-tracked) ----------
__device__ __forceinline__ void async_cp16(const unsigned short* g, unsigned short* l) {
#if __has_builtin(__builtin_amdgcn_global_load_async_to_lds_b128)
    __builtin_amdgcn_global_load_async_to_lds_b128(
        (__attribute__((address_space(1))) v4i_t*)(size_t)g,
        (__attribute__((address_space(3))) v4i_t*)(unsigned)(size_t)l,
        0, 0);
#else
    unsigned loff = (unsigned)(size_t)l;
    asm volatile("global_load_async_to_lds_b128 %0, %1, off"
                 :: "v"(loff), "v"(g) : "memory");
#endif
}
__device__ __forceinline__ void wait_async1() {
#if __has_builtin(__builtin_amdgcn_s_wait_asynccnt)
    __builtin_amdgcn_s_wait_asynccnt(1);
#else
    asm volatile("s_wait_asynccnt 1" ::: "memory");
#endif
}
__device__ __forceinline__ void wait_async0() {
#if __has_builtin(__builtin_amdgcn_s_wait_asynccnt)
    __builtin_amdgcn_s_wait_asynccnt(0);
#else
    asm volatile("s_wait_asynccnt 0" ::: "memory");
#endif
}

// ---------- generic fill ----------
__global__ void k_fill_u32(unsigned* p, unsigned v, long long n) {
    long long i = (long long)blockIdx.x * blockDim.x + threadIdx.x;
    if (i < n) p[i] = v;
}

// ---------- f32 -> bf16 elementwise ----------
__global__ void k_f32_to_bf16(const float* s, unsigned short* d, long long n) {
    long long i = (long long)blockIdx.x * blockDim.x + threadIdx.x;
    if (i < n) d[i] = f2bf(s[i]);
}

// ---------- W[K,NcReal] row-major f32 -> Wt[NcPad,Kpad] bf16 (column-major, zero-padded) ----------
__global__ void k_transpose_bf16(const float* W, unsigned short* Wt,
                                 int K, int NcPad, int NcReal, int Kpad) {
    long long i = (long long)blockIdx.x * blockDim.x + threadIdx.x;
    long long total = (long long)NcPad * Kpad;
    if (i >= total) return;
    int n = (int)(i / Kpad);
    int k = (int)(i % Kpad);
    float v = (k < K && n < NcReal) ? W[(long long)k * NcReal + n] : 0.0f;
    Wt[i] = f2bf(v);
}

// =====================================================================
// WMMA bf16 GEMM with async B-panel staging through LDS.
//   D[M,Npad] = A[M,K](row-major, pitch lda) * B(pre-transposed [Npad,Kpad], pitch ldb)
// Block = 256 threads = 8 waves; each wave owns a 32x64 output tile
// (2 M-subtiles x 4 N-subtiles = 8 f32 accumulators). The 64-column B panel
// (4KB per 32-k step) is shared by the whole block: staged into LDS with
// global_load_async_to_lds_b128, double-buffered, s_wait_asynccnt + barrier.
// Requires: K % 32 == 0, Npad % 64 == 0. M-tail handled by clamp+guard.
// =====================================================================
__global__ __launch_bounds__(256)
void k_gemm_wmma(const unsigned short* __restrict__ A,
                 const unsigned short* __restrict__ B,
                 unsigned short* Dbf, float* Df,
                 int M, int Npad, int K, int lda, int ldb, int ldd, int Nblocks) {
    __shared__ unsigned short smem[2][2048];     // 2 x 4KB B panels

    int w    = threadIdx.x >> 5;                 // wave 0..7
    int lane = threadIdx.x & 31;
    int l15  = lane & 15;
    int khalf = lane >> 4;

    int mb = blockIdx.x / Nblocks;
    int nb = blockIdx.x % Nblocks;
    int mbase = mb * 256 + w * 32;               // this wave's 32 rows
    int nbase = nb * 64;                         // block's 64 cols

    // A fragment row pointers (clamped for the M tail; stores are guarded)
    int r0 = mbase + l15;       if (r0 >= M) r0 = M - 1;
    int r1 = mbase + 16 + l15;  if (r1 >= M) r1 = M - 1;
    const unsigned short* a0p = A + (long long)r0 * lda + khalf * 16;
    const unsigned short* a1p = A + (long long)r1 * lda + khalf * 16;

    // async-copy mapping: thread t copies 16B: column t/4, chunk t%4
    int tcol = threadIdx.x >> 2;                 // 0..63
    int tchk = threadIdx.x & 3;                  // 0..3
    const unsigned short* gbase = B + (long long)(nbase + tcol) * ldb + tchk * 8;
    int lelem = tcol * 32 + tchk * 8;            // element offset inside a panel

    v8f acc[8];
    for (int i = 0; i < 8; ++i) acc[i] = v8f{};

    int nsteps = K / 32;
    async_cp16(gbase, &smem[0][lelem]);          // prologue: stage k-step 0

    for (int step = 0; step < nsteps; ++step) {
        int cur = step & 1;
        if (step + 1 < nsteps) {
            async_cp16(gbase + (step + 1) * 32, &smem[cur ^ 1][lelem]);
            wait_async1();                       // oldest (current panel) landed
        } else {
            wait_async0();
        }
        __syncthreads();                         // panel visible block-wide

        // A fragments: two 16B chunks per lane per 16-row subtile
        const unsigned short* ap0 = a0p + step * 32;
        const unsigned short* ap1 = a1p + step * 32;
        uint4 x0 = *(const uint4*)(ap0);
        uint4 x1 = *(const uint4*)(ap0 + 8);
        uint4 y0 = *(const uint4*)(ap1);
        uint4 y1 = *(const uint4*)(ap1 + 8);
        v16bf af0, af1;
        __builtin_memcpy(&af0, &x0, 16);
        __builtin_memcpy(((char*)&af0) + 16, &x1, 16);
        __builtin_memcpy(&af1, &y0, 16);
        __builtin_memcpy(((char*)&af1) + 16, &y1, 16);

        // 4 B fragments from LDS, 8 WMMAs
        #pragma unroll
        for (int ni = 0; ni < 4; ++ni) {
            const unsigned short* bp = &smem[cur][(ni * 16 + l15) * 32 + khalf * 16];
            v16bf bf = *(const v16bf*)bp;
            acc[ni] = __builtin_amdgcn_wmma_f32_16x16x32_bf16(
                false, af0, false, bf, (short)0, acc[ni], false, false);
            acc[4 + ni] = __builtin_amdgcn_wmma_f32_16x16x32_bf16(
                false, af1, false, bf, (short)0, acc[4 + ni], false, false);
        }
        __syncthreads();                         // done reading before overwrite
    }

    // store: D layout per ISA (VGPR r -> row r + 8*khalf within subtile)
    #pragma unroll
    for (int mi = 0; mi < 2; ++mi) {
        #pragma unroll
        for (int ni = 0; ni < 4; ++ni) {
            int dcol = nbase + ni * 16 + l15;
            #pragma unroll
            for (int r = 0; r < 8; ++r) {
                int drow = mbase + mi * 16 + r + khalf * 8;
                if (drow < M) {
                    float v = acc[mi * 4 + ni][r];
                    if (Dbf != nullptr) Dbf[(long long)drow * ldd + dcol] = f2bf(v);
                    else                Df [(long long)drow * ldd + dcol] = v;
                }
            }
        }
    }
}

// ---------- layer-1 per-(node,head) attention dots (bf16 features, pitch ldh) ----------
__global__ void k_alpha1(const unsigned short* __restrict__ h,
                         const float* __restrict__ asrc, const float* __restrict__ adst,
                         float* as_, float* ad_, int N, int ldh, int H, int C) {
    long long i = (long long)blockIdx.x * blockDim.x + threadIdx.x;
    if (i >= (long long)N * H) return;
    int n  = (int)(i / H);
    int hh = (int)(i % H);
    const unsigned short* hp = h + (long long)n * ldh + hh * C;
    const float* ps = asrc + hh * C;
    const float* pd = adst + hh * C;
    float ss = 0.f, sd = 0.f;
    for (int c = 0; c < C; ++c) { float v = bf2f(hp[c]); ss += v * ps[c]; sd += v * pd[c]; }
    as_[i] = ss; ad_[i] = sd;
}

// ---------- layer-2 per-node attention dots (f32 features, H=1) ----------
__global__ void k_alpha2(const float* __restrict__ h2,
                         const float* __restrict__ asrc, const float* __restrict__ adst,
                         float* as_, float* ad_, int N, int C) {
    int n = blockIdx.x * blockDim.x + threadIdx.x;
    if (n >= N) return;
    const float* hp = h2 + (long long)n * C;
    float ss = 0.f, sd = 0.f;
    for (int c = 0; c < C; ++c) { float v = hp[c]; ss += v * asrc[c]; sd += v * adst[c]; }
    as_[n] = ss; ad_[n] = sd;
}

// ---------- edge logits + segment max (mapped uint atomicMax) ----------
__global__ void k_edge_logits(const float* __restrict__ as_, const float* __restrict__ ad_,
                              const int* sA, const int* dA,
                              float* e, unsigned* m,
                              long long E, long long ET, int H) {
    long long i = (long long)blockIdx.x * blockDim.x + threadIdx.x;
    if (i >= ET * H) return;
    long long ed = i / H;
    int hh = (int)(i % H);
    int s, d; edge_sd(sA, dA, ed, E, s, d);
    float v = as_[(long long)s * H + hh] + ad_[(long long)d * H + hh];
    v = (v > 0.f) ? v : 0.2f * v;             // leaky_relu(0.2)
    e[i] = v;
    atomicMax(&m[(long long)d * H + hh], fmap(v));
}

// ---------- exp(e - max) in place + segment sum ----------
__global__ void k_edge_expsum(float* e, const unsigned* __restrict__ m, float* dn,
                              const int* sA, const int* dA,
                              long long E, long long ET, int H) {
    long long i = (long long)blockIdx.x * blockDim.x + threadIdx.x;
    if (i >= ET * H) return;
    long long ed = i / H;
    int hh = (int)(i % H);
    int s, d; edge_sd(sA, dA, ed, E, s, d);
    float ex = __expf(e[i] - funmap(m[(long long)d * H + hh]));
    e[i] = ex;
    atomicAdd(&dn[(long long)d * H + hh], ex);
}

// ---------- layer-1 weighted scatter-aggregate (bf16 src feats, pitch ldh -> f32 atomics) ----------
__global__ void k_agg1(const float* __restrict__ ex, const float* __restrict__ dn,
                       const unsigned short* __restrict__ h, float* out,
                       const int* sA, const int* dA,
                       long long E, long long ET, int ldh, int H, int C) {
    long long i = (long long)blockIdx.x * blockDim.x + threadIdx.x;
    if (i >= ET * H) return;
    long long ed = i / H;
    int hh = (int)(i % H);
    int s, d; edge_sd(sA, dA, ed, E, s, d);
    float coef = ex[i] / (dn[(long long)d * H + hh] + 1e-16f);
    const unsigned short* hp = h + (long long)s * ldh + hh * C;
    float* op = out + (long long)d * (H * C) + hh * C;
    for (int c = 0; c < C; ++c) atomicAdd(&op[c], coef * bf2f(hp[c]));
}

// ---------- layer-2 weighted scatter-aggregate (H=1, thread per (edge, channel)) ----------
__global__ void k_agg2(const float* __restrict__ ex, const float* __restrict__ dn,
                       const float* __restrict__ h2, float* out,
                       const int* sA, const int* dA,
                       long long E, long long ET, int C) {
    long long i = (long long)blockIdx.x * blockDim.x + threadIdx.x;
    if (i >= ET * C) return;
    long long ed = i / C;
    int c = (int)(i % C);
    int s, d; edge_sd(sA, dA, ed, E, s, d);
    float coef = ex[ed] / (dn[d] + 1e-16f);
    atomicAdd(&out[(long long)d * C + c], coef * h2[(long long)s * C + c]);
}

// ---------- bias + relu + pad-to-Kpad, emit bf16 activations for GEMM2 ----------
__global__ void k_bias_relu_pad(const float* __restrict__ acc, const float* __restrict__ b,
                                unsigned short* hbf, int N, int HC, int HCpad) {
    long long i = (long long)blockIdx.x * blockDim.x + threadIdx.x;
    if (i >= (long long)N * HCpad) return;
    int n = (int)(i / HCpad);
    int j = (int)(i % HCpad);
    float v = 0.f;
    if (j < HC) { v = acc[(long long)n * HC + j] + b[j]; v = (v > 0.f) ? v : 0.f; }
    hbf[i] = f2bf(v);
}

// ---------- final bias + relu in place on d_out ----------
__global__ void k_bias_relu(float* out, const float* __restrict__ b, int N, int C) {
    long long i = (long long)blockIdx.x * blockDim.x + threadIdx.x;
    if (i >= (long long)N * C) return;
    int c = (int)(i % C);
    float v = out[i] + b[c];
    out[i] = (v > 0.f) ? v : 0.f;
}

// =====================================================================
extern "C" void kernel_launch(void* const* d_in, const int* in_sizes, int n_in,
                              void* d_out, int out_size, void* d_ws, size_t ws_size,
                              hipStream_t stream) {
    const int IN = 128, H1 = 36, C1 = 36, HC = 1296, HCpad = 1312, OUT = 128;
    const int NPAD1 = 1344;                         // 1296 padded to 64-col blocks
    const int N = in_sizes[0] / IN;                 // 20000
    const long long E  = in_sizes[1] / 2;           // 160000
    const long long ET = E + N;                     // +self-loops

    const float* x      = (const float*)d_in[0];
    const int*   eidx   = (const int*)d_in[1];
    const float* W1     = (const float*)d_in[2];
    const float* asrc1  = (const float*)d_in[3];
    const float* adst1  = (const float*)d_in[4];
    const float* b1     = (const float*)d_in[5];
    const float* W2     = (const float*)d_in[6];
    const float* asrc2  = (const float*)d_in[7];
    const float* adst2  = (const float*)d_in[8];
    const float* b2     = (const float*)d_in[9];
    const int* srcA = eidx;
    const int* dstA = eidx + E;
    float* outp = (float*)d_out;

    // ---- workspace carve-out ----
    char* ws = (char*)d_ws; size_t off = 0;
    auto carve = [&](size_t bytes) -> void* {
        void* p = ws + off; off = (off + bytes + 255) & ~(size_t)255; return p;
    };
    unsigned short* Xbf  = (unsigned short*)carve((size_t)N * IN * 2);
    unsigned short* W1t  = (unsigned short*)carve((size_t)NPAD1 * IN * 2);    // [1344,128]
    unsigned short* W2t  = (unsigned short*)carve((size_t)OUT * HCpad * 2);   // [128,1312]
    unsigned short* h1bf = (unsigned short*)carve((size_t)N * NPAD1 * 2);     // GEMM1 out (bf16, pitch 1344)
    float*    as1 = (float*)carve((size_t)N * H1 * 4);
    float*    ad1 = (float*)carve((size_t)N * H1 * 4);
    unsigned* m1  = (unsigned*)carve((size_t)N * H1 * 4);
    float*    dn1 = (float*)carve((size_t)N * H1 * 4);
    float*    e1  = (float*)carve((size_t)ET * H1 * 4);
    float*    acc1 = (float*)carve((size_t)N * HC * 4);                       // f32 aggregate (L2-resident)
    unsigned short* H1bf = (unsigned short*)carve((size_t)N * HCpad * 2);     // relu(out1+b1), K-padded
    float*    h2  = (float*)carve((size_t)N * OUT * 4);
    float*    as2 = (float*)carve((size_t)N * 4);
    float*    ad2 = (float*)carve((size_t)N * 4);
    unsigned* m2  = (unsigned*)carve((size_t)N * 4);
    float*    dn2 = (float*)carve((size_t)N * 4);
    float*    e2  = (float*)carve((size_t)ET * 4);
    (void)ws_size; (void)n_in; (void)out_size;

    const int B = 256;
    auto nb = [&](long long n) { return (unsigned)((n + B - 1) / B); };
    const unsigned NEG_INF_MAP = 0x007FFFFFu;       // fmap(-inf)

    // ---- precision conversion + weight transposes ----
    k_f32_to_bf16<<<nb((long long)N * IN), B, 0, stream>>>(x, Xbf, (long long)N * IN);
    k_transpose_bf16<<<nb((long long)NPAD1 * IN), B, 0, stream>>>(W1, W1t, IN, NPAD1, HC, IN);
    k_transpose_bf16<<<nb((long long)OUT * HCpad), B, 0, stream>>>(W2, W2t, HC, OUT, OUT, HCpad);

    // ---- GEMM1: h1 = X @ W1  (bf16 WMMA, async LDS staging, bf16 out) ----
    {
        int Mblocks = (N + 255) / 256;               // 79
        int Nblocks = NPAD1 / 64;                    // 21
        k_gemm_wmma<<<Mblocks * Nblocks, 256, 0, stream>>>(
            Xbf, W1t, h1bf, nullptr, N, NPAD1, IN, IN, IN, NPAD1, Nblocks);
    }

    // ---- layer-1 attention ----
    k_alpha1<<<nb((long long)N * H1), B, 0, stream>>>(h1bf, asrc1, adst1, as1, ad1, N, NPAD1, H1, C1);
    k_fill_u32<<<nb((long long)N * H1), B, 0, stream>>>(m1, NEG_INF_MAP, (long long)N * H1);
    k_fill_u32<<<nb((long long)N * H1), B, 0, stream>>>((unsigned*)dn1, 0u, (long long)N * H1);
    k_fill_u32<<<nb((long long)N * HC), B, 0, stream>>>((unsigned*)acc1, 0u, (long long)N * HC);
    k_edge_logits<<<nb(ET * H1), B, 0, stream>>>(as1, ad1, srcA, dstA, e1, m1, E, ET, H1);
    k_edge_expsum<<<nb(ET * H1), B, 0, stream>>>(e1, m1, dn1, srcA, dstA, E, ET, H1);
    k_agg1<<<nb(ET * H1), B, 0, stream>>>(e1, dn1, h1bf, acc1, srcA, dstA, E, ET, NPAD1, H1, C1);
    k_bias_relu_pad<<<nb((long long)N * HCpad), B, 0, stream>>>(acc1, b1, H1bf, N, HC, HCpad);

    // ---- GEMM2: h2 = H1 @ W2  (bf16 WMMA, f32 out; K padded 1296->1312) ----
    {
        int Mblocks = (N + 255) / 256;               // 79
        int Nblocks = OUT / 64;                      // 2
        k_gemm_wmma<<<Mblocks * Nblocks, 256, 0, stream>>>(
            H1bf, W2t, nullptr, h2, N, OUT, HCpad, HCpad, HCpad, OUT, Nblocks);
    }

    // ---- layer-2 attention (H=1) ----
    k_alpha2<<<nb(N), B, 0, stream>>>(h2, asrc2, adst2, as2, ad2, N, OUT);
    k_fill_u32<<<nb(N), B, 0, stream>>>(m2, NEG_INF_MAP, N);
    k_fill_u32<<<nb(N), B, 0, stream>>>((unsigned*)dn2, 0u, N);
    k_fill_u32<<<nb((long long)N * OUT), B, 0, stream>>>((unsigned*)outp, 0u, (long long)N * OUT);
    k_edge_logits<<<nb(ET), B, 0, stream>>>(as2, ad2, srcA, dstA, e2, m2, E, ET, 1);
    k_edge_expsum<<<nb(ET), B, 0, stream>>>(e2, m2, dn2, srcA, dstA, E, ET, 1);
    k_agg2<<<nb(ET * OUT), B, 0, stream>>>(e2, dn2, h2, outp, srcA, dstA, E, ET, OUT);
    k_bias_relu<<<nb((long long)N * OUT), B, 0, stream>>>(outp, b2, N, OUT);
}
